// SELF_Former_77713138254512
// MI455X (gfx1250) — matrix operand order repaired and checked by reading, compiled, not run on hardware
//
#include <hip/hip_runtime.h>
#include <stdint.h>

typedef __attribute__((ext_vector_type(16))) __bf16 v16bf;
typedef __attribute__((ext_vector_type(8)))  float  v8f;
typedef __bf16 bf16;

#define TILE_M 128
#define TILE_N 128
#define KSTEP  32
#define LDST   40   // LDS row stride in bf16 elems (32 data + 8 pad), 80B rows (16B aligned)

union FragU { uint4 u[2]; v16bf v; };

// Workgroup barrier that only orders LDS traffic (DScnt), leaving the register
// prefetch global loads in flight across the barrier (unlike __syncthreads(),
// which waits loadcnt 0). CDNA5 split barrier: S_BARRIER_SIGNAL/WAIT -1.
__device__ __forceinline__ void wg_barrier_lds() {
  asm volatile("s_wait_dscnt 0x0\n\t"
               "s_barrier_signal -1\n\t"
               "s_barrier_wait -1" ::: "memory");
}

// A fragment (16x32 bf16): lane r=L&15 holds row M=r; half h=L>>4 selects K-groups
// VGPR0..3 = K in [h*8, h*8+8), VGPR4..7 = K in [16+h*8, 16+h*8+8)  (ISA 7.12.2)
__device__ __forceinline__ v16bf lds_frag_a(const bf16* tile, int lane) {
  int r = lane & 15, h = lane >> 4;
  const bf16* p = tile + r * LDST;
  FragU f;
  f.u[0] = *(const uint4*)(p + h * 8);
  f.u[1] = *(const uint4*)(p + 16 + h * 8);
  return f.v;
}
// B fragment (32x16 bf16, stored as W rows [n][k]): lane r holds col N=r;
// same K layout as A: VGPR0..3 = K in [h*8, h*8+8), VGPR4..7 = K in [16+h*8, 16+h*8+8)
__device__ __forceinline__ v16bf lds_frag_b(const bf16* tile, int lane) {
  int r = lane & 15, h = lane >> 4;
  const bf16* p = tile + r * LDST;
  FragU f;
  f.u[0] = *(const uint4*)(p + h * 8);
  f.u[1] = *(const uint4*)(p + 16 + h * 8);
  return f.v;
}

__device__ __forceinline__ int imin(int a, int b) { return a < b ? a : b; }

// C = A * B^T (+bias, optional leaky relu). A:[M x K] bf16 (lda), B:[N x K] bf16 (ldb).
// Requires K % 32 == 0. Out-of-range staged rows are clamped (their products only feed
// output rows/cols >= M/N, which only the GUARD epilogue can skip; GUARD=false is only
// legal when M % 128 == 0 and N % 128 == 0).
// OMODE: 0 = bf16 C[m][n], 1 = f32 C[m][n], 2 = f32 transposed C[n][m]
template <int OMODE, bool RELU, bool GUARD>
__global__ void __launch_bounds__(256)
gemm_wmma(const bf16* __restrict__ A, long lda,
          const bf16* __restrict__ B, long ldb,
          const float* __restrict__ bias,
          void* __restrict__ Cout, long ldc,
          int M, int N, int K)
{
  __shared__ bf16 At[TILE_M * LDST];
  __shared__ bf16 Bt[TILE_N * LDST];

  int tid  = threadIdx.x;
  int lane = tid & 31;
  int wave = tid >> 5;       // 0..7
  int wm   = wave & 1;       // 2 waves along M: rows wm*64..wm*64+63
  int wn   = wave >> 1;      // 4 waves along N: cols wn*32..wn*32+31
  int m0   = blockIdx.y * TILE_M;
  int n0   = blockIdx.x * TILE_N;

  // Branch-free staging: each thread owns 2 chunks (8 bf16 each) per tile.
  int c0 = tid, c1 = tid + 256;
  int r0 = c0 >> 2, col0 = (c0 & 3) * 8;
  int r1 = c1 >> 2, col1 = (c1 & 3) * 8;
  const bf16* pa0 = A + (long)imin(m0 + r0, M - 1) * lda + col0;
  const bf16* pa1 = A + (long)imin(m0 + r1, M - 1) * lda + col1;
  const bf16* pb0 = B + (long)imin(n0 + r0, N - 1) * ldb + col0;
  const bf16* pb1 = B + (long)imin(n0 + r1, N - 1) * ldb + col1;
  bf16* sa0 = At + r0 * LDST + col0;
  bf16* sa1 = At + r1 * LDST + col1;
  bf16* sb0 = Bt + r0 * LDST + col0;
  bf16* sb1 = Bt + r1 * LDST + col1;

  v8f acc[4][2];
  const v8f zero = {0.f, 0.f, 0.f, 0.f, 0.f, 0.f, 0.f, 0.f};
#pragma unroll
  for (int t = 0; t < 4; ++t)
#pragma unroll
    for (int u = 0; u < 2; ++u) acc[t][u] = zero;

  // Prologue: tile 0 into registers.
  uint4 a0 = *(const uint4*)pa0;
  uint4 a1 = *(const uint4*)pa1;
  uint4 b0 = *(const uint4*)pb0;
  uint4 b1 = *(const uint4*)pb1;

  for (int k0 = 0; k0 < K; k0 += KSTEP) {
    wg_barrier_lds();                // all waves done reading previous tile
    *(uint4*)sa0 = a0; *(uint4*)sa1 = a1;
    *(uint4*)sb0 = b0; *(uint4*)sb1 = b1;
    // Prefetch tile i+1 (clamped re-load of current tile on the last iteration:
    // branch-free; the value is never stored). These loads stay outstanding across
    // the barrier and are only waited at the next iteration's ds_store.
    int kn = (k0 + KSTEP < K) ? (k0 + KSTEP) : k0;
    a0 = *(const uint4*)(pa0 + kn);
    a1 = *(const uint4*)(pa1 + kn);
    b0 = *(const uint4*)(pb0 + kn);
    b1 = *(const uint4*)(pb1 + kn);
    wg_barrier_lds();                // tile i visible in LDS

    v16bf bfr[2];
#pragma unroll
    for (int u = 0; u < 2; ++u)
      bfr[u] = lds_frag_b(Bt + (wn * 32 + u * 16) * LDST, lane);
#pragma unroll
    for (int t = 0; t < 4; ++t) {
      v16bf afr = lds_frag_a(At + (wm * 64 + t * 16) * LDST, lane);
#pragma unroll
      for (int u = 0; u < 2; ++u)
        acc[t][u] = __builtin_amdgcn_wmma_f32_16x16x32_bf16(
            false, afr, false, bfr[u], (short)0, acc[t][u], false, false);
    }
  }

  // Epilogue. C/D layout: lane n = L&15, VGPR v -> m = v + 8*(L>>4)
  int lr = lane & 15, h = lane >> 4;
#pragma unroll
  for (int t = 0; t < 4; ++t) {
#pragma unroll
    for (int u = 0; u < 2; ++u) {
      int n = n0 + wn * 32 + u * 16 + lr;
      float bv = 0.f;
      if (bias != nullptr && (!GUARD || n < N)) bv = bias[n];
#pragma unroll
      for (int v = 0; v < 8; ++v) {
        int m = m0 + wm * 64 + t * 16 + v + 8 * h;
        if (!GUARD || (m < M && n < N)) {
          float c = acc[t][u][v] + bv;
          if (RELU) c = (c > 0.f) ? c : 0.2f * c;
          if (OMODE == 0)       ((bf16*)Cout)[(long)m * ldc + n] = (bf16)c;
          else if (OMODE == 1)  ((float*)Cout)[(long)m * ldc + n] = c;
          else                  ((float*)Cout)[(long)n * ldc + m] = c;
        }
      }
    }
  }
}

__global__ void __launch_bounds__(256)
softmax_rows(const float* __restrict__ S, bf16* __restrict__ P, int ncols, float scale) {
  int row = blockIdx.x;
  const float* s = S + (long)row * ncols;
  bf16* p = P + (long)row * ncols;
  __shared__ float red[256];
  int tid = threadIdx.x;

  float mx = -3.4e38f;
  for (int j = tid; j < ncols; j += 256) mx = fmaxf(mx, s[j] * scale);
  red[tid] = mx; __syncthreads();
  for (int w = 128; w > 0; w >>= 1) { if (tid < w) red[tid] = fmaxf(red[tid], red[tid + w]); __syncthreads(); }
  mx = red[0]; __syncthreads();

  float sum = 0.f;
  for (int j = tid; j < ncols; j += 256) sum += __expf(s[j] * scale - mx);
  red[tid] = sum; __syncthreads();
  for (int w = 128; w > 0; w >>= 1) { if (tid < w) red[tid] += red[tid + w]; __syncthreads(); }
  float inv = 1.0f / red[0];

  for (int j = tid; j < ncols; j += 256) p[j] = (bf16)(__expf(s[j] * scale - mx) * inv);
}

__global__ void diag_extract(const bf16* __restrict__ P, long ld, float* __restrict__ diag, int n) {
  int i = blockIdx.x * 256 + threadIdx.x;
  if (i < n) diag[i] = (float)P[(long)i * ld + i];
}

// rank[i] = #{ j : d[j] > d[i] or (d[j]==d[i] and j<i) }; selected iff rank < k.
// (Row order of `sel` is irrelevant: softmax+sum over selected rows is permutation-invariant.)
__global__ void __launch_bounds__(256)
topk_select(const float* __restrict__ diag, int n, int k, int* __restrict__ sel) {
  int i = blockIdx.x;
  float di = diag[i];
  __shared__ int red[256];
  int c = 0;
  for (int j = threadIdx.x; j < n; j += 256) {
    float dj = diag[j];
    if (dj > di || (dj == di && j < i)) c++;
  }
  red[threadIdx.x] = c; __syncthreads();
  for (int w = 128; w > 0; w >>= 1) { if (threadIdx.x < w) red[threadIdx.x] += red[threadIdx.x + w]; __syncthreads(); }
  if (threadIdx.x == 0) { int r = red[0]; if (r < k) sel[r] = i; }
}

__global__ void gather_add(const bf16* __restrict__ a, const bf16* __restrict__ b,
                           const int* __restrict__ sel, bf16* __restrict__ out, int cols) {
  int c = blockIdx.x * 256 + threadIdx.x;
  int r = blockIdx.y;
  if (c < cols) {
    int src = sel[r];
    out[(long)r * cols + c] = (bf16)((float)a[(long)src * cols + c] + (float)b[(long)src * cols + c]);
  }
}

// in [R x C] bf16 -> out [C x R] bf16
__global__ void __launch_bounds__(256)
transpose_bf16(const bf16* __restrict__ in, bf16* __restrict__ out, int R, int C) {
  __shared__ bf16 t[32][33];
  int bx = blockIdx.x * 32, by = blockIdx.y * 32;
  int lx = threadIdx.x & 31, ly = threadIdx.x >> 5;
#pragma unroll
  for (int i = 0; i < 32; i += 8) {
    int r = by + ly + i, c = bx + lx;
    t[ly + i][lx] = (r < R && c < C) ? in[(long)r * C + c] : (bf16)0.f;
  }
  __syncthreads();
#pragma unroll
  for (int i = 0; i < 32; i += 8) {
    int r = bx + ly + i, c = by + lx;
    if (r < C && c < R) out[(long)r * R + c] = t[lx][ly + i];
  }
}

// in [R x C] f32 -> out [C x R] bf16 with output leading dim ldo (for x.T with K-pad)
__global__ void __launch_bounds__(256)
transpose_cvt_f32_bf16(const float* __restrict__ in, bf16* __restrict__ out, int R, int C, long ldo) {
  __shared__ float t[32][33];
  int bx = blockIdx.x * 32, by = blockIdx.y * 32;
  int lx = threadIdx.x & 31, ly = threadIdx.x >> 5;
#pragma unroll
  for (int i = 0; i < 32; i += 8) {
    int r = by + ly + i, c = bx + lx;
    t[ly + i][lx] = (r < R && c < C) ? in[(long)r * C + c] : 0.f;
  }
  __syncthreads();
#pragma unroll
  for (int i = 0; i < 32; i += 8) {
    int r = bx + ly + i, c = by + lx;
    if (r < C && c < R) out[(long)r * ldo + c] = (bf16)t[lx][ly + i];
  }
}

// f32 [rows x cols] -> bf16 [rows x ldo] (pad columns untouched; zero buffer first if ldo > cols)
__global__ void __launch_bounds__(256)
cvt_f32_bf16_strided(const float* __restrict__ in, bf16* __restrict__ out,
                     int rows, int cols, long ldo) {
  long i = (long)blockIdx.x * 256 + threadIdx.x;
  if (i < (long)rows * cols) {
    long r = i / cols, c = i - r * cols;
    out[r * ldo + c] = (bf16)in[i];
  }
}

__global__ void __launch_bounds__(256)
zero_bf16(bf16* __restrict__ p, long n) {
  long i = (long)blockIdx.x * 256 + threadIdx.x;
  if (i < n) p[i] = (bf16)0.f;
}

#define GEMM(OM, RELU, GUARD, A, LDA, B, LDB, BIAS, OUT, LDC, M, NN, K)                        \
  gemm_wmma<OM, RELU, GUARD><<<dim3(((NN) + 127) / 128, ((M) + 127) / 128), 256, 0, stream>>>( \
      (const bf16*)(A), (long)(LDA), (const bf16*)(B), (long)(LDB), (BIAS),                    \
      (void*)(OUT), (long)(LDC), (M), (NN), (K))

extern "C" void kernel_launch(void* const* d_in, const int* in_sizes, int n_in,
                              void* d_out, int out_size, void* d_ws, size_t ws_size,
                              hipStream_t stream) {
  (void)in_sizes; (void)n_in; (void)out_size; (void)ws_size;
  const int N = 4096, S2 = 3000, S1 = 2000, D1 = 2048, D2 = 1024;
  const int S2P = 3008;  // S2 padded to a multiple of 32 (zero pad)
  enum { E1 = 0, E2, L1, L2, L3, Q1, K1, V1, Q2, K2, V2, Q3, K3, V3, Q4, K4, V4 };
  static const int DOUTa[17] = {2048,1024,1024,1024,2000,1024,1024,1024,1024,1024,1024,1024,1024,1024,1024,1024,1024};
  static const int DINa[17]  = {3000,2048,1024,3072,1024,1024,1024,1024,1024,1024,1024,1024,1024,1024,1024,1024,1024};

  const float* x = (const float*)d_in[0];
  const float* Wf[17]; const float* Bf[17];
  for (int i = 0; i < 17; ++i) { Wf[i] = (const float*)d_in[1 + 2 * i]; Bf[i] = (const float*)d_in[2 + 2 * i]; }

  char* base = (char*)d_ws; size_t off = 0;
  auto alloc = [&](size_t bytes) -> void* {
    off = (off + 255) & ~(size_t)255;
    void* p = base + off; off += bytes; return p;
  };

  // bf16 weight copies (deterministic: regenerated each call). e1 gets K-padded layout.
  bf16* Wb[17];
  for (int i = 0; i < 17; ++i) {
    long ldo = (i == E1) ? S2P : DINa[i];
    long nel = (long)DOUTa[i] * ldo;
    long nsrc = (long)DOUTa[i] * DINa[i];
    Wb[i] = (bf16*)alloc((size_t)nel * 2);
    if (i == E1)
      zero_bf16<<<dim3((unsigned)((nel + 255) / 256)), 256, 0, stream>>>(Wb[i], nel);
    cvt_f32_bf16_strided<<<dim3((unsigned)((nsrc + 255) / 256)), 256, 0, stream>>>(
        Wf[i], Wb[i], DOUTa[i], DINa[i], ldo);
  }

  bf16* xT    = (bf16*)alloc((size_t)N * S2P * 2);
  bf16* lat1  = (bf16*)alloc((size_t)N * D1 * 2);
  bf16* lat   = (bf16*)alloc((size_t)N * D2 * 2);
  bf16* latst = (bf16*)alloc((size_t)N * D2 * 2);
  bf16* qb    = (bf16*)alloc((size_t)N * D2 * 2);
  bf16* kb    = (bf16*)alloc((size_t)N * D2 * 2);
  bf16* vb    = (bf16*)alloc((size_t)N * D2 * 2);
  bf16* vT    = (bf16*)alloc((size_t)N * D2 * 2);
  float* S    = (float*)alloc((size_t)N * N * 4);
  bf16* P     = (bf16*)alloc((size_t)N * N * 2);
  bf16* cat   = (bf16*)alloc((size_t)N * 3 * D2 * 2);
  bf16* out1  = (bf16*)alloc((size_t)N * D2 * 2);
  bf16* osel  = (bf16*)alloc((size_t)(N / 2) * D2 * 2);
  bf16* lst2  = (bf16*)alloc((size_t)N * D2 * 2);
  float* diag = (float*)alloc((size_t)N * 4);
  int* sel    = (int*)alloc((size_t)(N / 2) * 4);

  const float scale = 1.0f / 32.0f;  // 1/sqrt(D2)

  // x.T : [S2 x N] f32 -> [N x S2P] bf16 (pad columns zeroed)
  {
    long nel = (long)N * S2P;
    zero_bf16<<<dim3((unsigned)((nel + 255) / 256)), 256, 0, stream>>>(xT, nel);
  }
  transpose_cvt_f32_bf16<<<dim3((N + 31) / 32, (S2 + 31) / 32), 256, 0, stream>>>(x, xT, S2, N, S2P);

  // encoder
  GEMM(0, true, false, xT,   S2P, Wb[E1], S2P, Bf[E1], lat1, D1, N, D1, S2P);
  GEMM(0, true, false, lat1, D1,  Wb[E2], D1,  Bf[E2], lat,  D2, N, D2, D1);

  // ---- attn1 (Q2,K2,V2) -> cat[:, D2:2*D2] ----
  GEMM(0, true, false, lat, D2, Wb[Q2], D2, Bf[Q2], qb, D2, N, D2, D2);
  GEMM(0, true, false, lat, D2, Wb[K2], D2, Bf[K2], kb, D2, N, D2, D2);
  GEMM(0, true, false, lat, D2, Wb[V2], D2, Bf[V2], vb, D2, N, D2, D2);
  GEMM(1, false, false, qb, D2, kb, D2, (const float*)nullptr, S, N, N, N, D2);
  softmax_rows<<<N, 256, 0, stream>>>(S, P, N, scale);
  transpose_bf16<<<dim3((D2 + 31) / 32, (N + 31) / 32), 256, 0, stream>>>(vb, vT, N, D2);
  GEMM(0, false, false, P, N, vT, N, (const float*)nullptr, cat + D2, 3 * D2, N, D2, N);

  // ---- attn2 (Q3,K3,V3) -> cat[:, 2*D2:3*D2] ----
  GEMM(0, true, false, lat, D2, Wb[Q3], D2, Bf[Q3], qb, D2, N, D2, D2);
  GEMM(0, true, false, lat, D2, Wb[K3], D2, Bf[K3], kb, D2, N, D2, D2);
  GEMM(0, true, false, lat, D2, Wb[V3], D2, Bf[V3], vb, D2, N, D2, D2);
  GEMM(1, false, false, qb, D2, kb, D2, (const float*)nullptr, S, N, N, N, D2);
  softmax_rows<<<N, 256, 0, stream>>>(S, P, N, scale);
  transpose_bf16<<<dim3((D2 + 31) / 32, (N + 31) / 32), 256, 0, stream>>>(vb, vT, N, D2);
  GEMM(0, false, false, P, N, vT, N, (const float*)nullptr, cat + 2 * D2, 3 * D2, N, D2, N);

  // ---- latent_st = lrelu(l1(latent)) ----
  GEMM(0, true, false, lat, D2, Wb[L1], D2, Bf[L1], latst, D2, N, D2, D2);

  // ---- self_trans, stage 1 (Q1,K1,V1) ----
  GEMM(0, true, false, latst, D2, Wb[Q1], D2, Bf[Q1], qb, D2, N, D2, D2);
  GEMM(0, true, false, latst, D2, Wb[K1], D2, Bf[K1], kb, D2, N, D2, D2);
  GEMM(0, true, false, latst, D2, Wb[V1], D2, Bf[V1], vb, D2, N, D2, D2);
  GEMM(1, false, false, qb, D2, kb, D2, (const float*)nullptr, S, N, N, N, D2);
  softmax_rows<<<N, 256, 0, stream>>>(S, P, N, scale);
  diag_extract<<<(N + 255) / 256, 256, 0, stream>>>(P, N, diag, N);
  topk_select<<<N, 256, 0, stream>>>(diag, N, N / 2, sel);
  transpose_bf16<<<dim3((D2 + 31) / 32, (N + 31) / 32), 256, 0, stream>>>(vb, vT, N, D2);
  GEMM(0, false, false, P, N, vT, N, (const float*)nullptr, out1, D2, N, D2, N);  // out = W @ v
  // out_select = out[sel] + latent_st[sel]
  gather_add<<<dim3(D2 / 256, N / 2), 256, 0, stream>>>(out1, latst, sel, osel, D2);

  // ---- self_trans, stage 2 (Q4 on out, K4/V4 on out_select) ----
  GEMM(0, true, false, out1, D2, Wb[Q4], D2, Bf[Q4], qb, D2, N,     D2, D2);
  GEMM(0, true, false, osel, D2, Wb[K4], D2, Bf[K4], kb, D2, N / 2, D2, D2);
  GEMM(0, true, false, osel, D2, Wb[V4], D2, Bf[V4], vb, D2, N / 2, D2, D2);
  GEMM(1, false, false, qb, D2, kb, D2, (const float*)nullptr, S, N / 2, N, N / 2, D2);
  softmax_rows<<<N, 256, 0, stream>>>(S, P, N / 2, scale);
  transpose_bf16<<<dim3((D2 + 31) / 32, (N / 2 + 31) / 32), 256, 0, stream>>>(vb, vT, N / 2, D2);
  GEMM(0, false, false, P, N / 2, vT, N / 2, (const float*)nullptr, cat + 0, 3 * D2, N, D2, N / 2);

  // ---- l2 + l3 (l3 stored transposed into d_out: [S1 x N] f32) ----
  GEMM(0, true, false, cat, 3 * D2, Wb[L2], 3 * D2, Bf[L2], lst2, D2, N, D2, 3 * D2);
  GEMM(2, false, true, lst2, D2, Wb[L3], D2, Bf[L3], d_out, N, N, S1, D2);
}
